// GroupedQueryAttention_55138790146626
// MI455X (gfx1250) — matrix operand order repaired and checked by reading, not compiled, not run on hardware
//
#include <hip/hip_runtime.h>

// ---------------------------------------------------------------- constants
#define B_   2
#define T_   2048
#define DIN  1024
#define DOUT 1024
#define H_   16
#define G_   4
#define HD   64          // head dim
#define NTOK (B_ * T_)   // 4096

typedef __bf16 v16bf __attribute__((ext_vector_type(16)));
typedef float  v8f   __attribute__((ext_vector_type(8)));
typedef unsigned int v4u __attribute__((ext_vector_type(4)));
typedef int v8i __attribute__((ext_vector_type(8)));
typedef int v4i __attribute__((ext_vector_type(4)));

union Frag16 {
    v16bf v;
    uint4 q[2];
    __bf16 e[16];
};

__device__ __forceinline__ void wave_lds_sync() {
    __builtin_amdgcn_wave_barrier();
    asm volatile("s_wait_dscnt 0" ::: "memory");
    __builtin_amdgcn_wave_barrier();
}

// ---- DPP16 butterfly reductions within each 16-lane half (VALU only) ----
#define DPP_F(x, ctrl) \
    __int_as_float(__builtin_amdgcn_mov_dpp(__float_as_int(x), (ctrl), 0xf, 0xf, true))

__device__ __forceinline__ float row_max16(float x) {
    x = fmaxf(x, DPP_F(x, 0xB1));   // quad_perm [1,0,3,2]
    x = fmaxf(x, DPP_F(x, 0x4E));   // quad_perm [2,3,0,1]
    x = fmaxf(x, DPP_F(x, 0x141));  // row_half_mirror
    x = fmaxf(x, DPP_F(x, 0x140));  // row_mirror
    return x;
}
__device__ __forceinline__ float row_sum16(float x) {
    x += DPP_F(x, 0xB1);
    x += DPP_F(x, 0x4E);
    x += DPP_F(x, 0x141);
    x += DPP_F(x, 0x140);
    return x;
}

// ---- Tensor Data Mover: 2D bf16 tile global->LDS (D# per CDNA5 ISA) ----
// 6-arg builtin form (clang-23 / therock headers): (v4u, v8i, v4i, v4i, v8i, i32 cpol)
__device__ __forceinline__ void tdm_load_2d_bf16(unsigned lds_off, const void* gaddr,
                                                 unsigned tile_d0, unsigned tile_d1,
                                                 unsigned tensor_d0, unsigned tensor_d1,
                                                 unsigned stride_d0) {
    const unsigned long long ga = (unsigned long long)(size_t)gaddr;
    v4u g0;
    g0[0] = 1u;                                   // count=1, user mode
    g0[1] = lds_off;                              // LDS byte address
    g0[2] = (unsigned)ga;                         // global addr [31:0]
    g0[3] = (unsigned)(ga >> 32) | 0x80000000u;   // global addr [56:32] | type=2
    v8i g1;
    g1[0] = (int)(1u << 16);                      // data_size = 1 (2 bytes)
    g1[1] = (int)((tensor_d0 & 0xFFFFu) << 16);   // tensor_dim0[15:0] @ bits 63:48
    g1[2] = (int)(((tensor_d0 >> 16) & 0xFFFFu) | ((tensor_d1 & 0xFFFFu) << 16));
    g1[3] = (int)(((tensor_d1 >> 16) & 0xFFFFu) | ((tile_d0 & 0xFFFFu) << 16));
    g1[4] = (int)(tile_d1 & 0xFFFFu);             // tile_dim1 (tile_dim2 = 0)
    g1[5] = (int)stride_d0;                       // tensor_dim0_stride[31:0]
    g1[6] = 0;
    g1[7] = 0;                                    // tensor_dim1_stride unused (2D tile)
    const v4i gz4 = {0, 0, 0, 0};
    const v8i gz8 = {0, 0, 0, 0, 0, 0, 0, 0};
    __builtin_amdgcn_tensor_load_to_lds(g0, g1, gz4, gz4, gz8, 0);
}

// ---------------------------------------------------------------- fp32 -> bf16
__global__ void f2bf_kernel(const float* __restrict__ src, __bf16* __restrict__ dst, int n) {
    int i = blockIdx.x * blockDim.x + threadIdx.x;
    if (i < n) dst[i] = (__bf16)src[i];
}

// fp32 (K,N) -> bf16 transposed (N,K)
__global__ void f2bf_t_kernel(const float* __restrict__ src, __bf16* __restrict__ dst,
                              int K, int N) {
    int i = blockIdx.x * blockDim.x + threadIdx.x;
    if (i >= K * N) return;
    int k = i / N, n = i % N;
    dst[(size_t)n * K + k] = (__bf16)src[i];
}

// bf16 V (B*T, G*HD) -> VT (B, G, HD, T)
__global__ void vtrans_kernel(const __bf16* __restrict__ V, __bf16* __restrict__ VT) {
    int i = blockIdx.x * blockDim.x + threadIdx.x;
    if (i >= NTOK * G_ * HD) return;
    int tok = i / (G_ * HD), r = i % (G_ * HD);
    int g = r / HD, d = r % HD;
    int b = tok / T_, t = tok % T_;
    VT[(((size_t)b * G_ + g) * HD + d) * T_ + t] = V[i];
}

// ---------------------------------------------------------------- WMMA GEMM (B transposed)
// C(M,N) = A(M,K) @ BT(N,K)^T ; one 16x64 C tile per wave.
template <bool F32OUT>
__global__ void gemm_wmma_nt(const __bf16* __restrict__ A, const __bf16* __restrict__ BT,
                             void* __restrict__ C, int M, int N, int K) {
    const int wave   = (int)((blockIdx.x * blockDim.x + threadIdx.x) >> 5);
    const int lane   = threadIdx.x & 31;
    const int tilesN = N >> 6;
    if (wave >= (M >> 4) * tilesN) return;
    const int tm = wave / tilesN, tn = wave % tilesN;
    const int m  = lane & 15, h = lane >> 4;

    const __bf16* arow = A + (size_t)(tm * 16 + m) * K;
    const __bf16* brow[4];
#pragma unroll
    for (int j = 0; j < 4; ++j)
        brow[j] = BT + (size_t)(tn * 64 + j * 16 + m) * K;

    const v8f vzero = {};
    v8f acc[4];
#pragma unroll
    for (int j = 0; j < 4; ++j) acc[j] = vzero;

    for (int kb = 0; kb < K; kb += 32) {
        Frag16 a;
        a.q[0] = *(const uint4*)(arow + kb + h * 8);
        a.q[1] = *(const uint4*)(arow + kb + 16 + h * 8);
        __builtin_prefetch(arow + kb + 256, 0, 1);
#pragma unroll
        for (int j = 0; j < 4; ++j) {
            Frag16 b;
            b.q[0] = *(const uint4*)(brow[j] + kb + h * 8);
            b.q[1] = *(const uint4*)(brow[j] + kb + 16 + h * 8);
            acc[j] = __builtin_amdgcn_wmma_f32_16x16x32_bf16(false, a.v, false, b.v,
                                                             (short)0, acc[j], false, false);
        }
    }
#pragma unroll
    for (int j = 0; j < 4; ++j) {
        const int col = tn * 64 + j * 16 + m;
#pragma unroll
        for (int r = 0; r < 8; ++r) {
            const size_t idx = (size_t)(tm * 16 + r + 8 * h) * N + col;
            if (F32OUT) ((float*)C)[idx] = acc[j][r];
            else        ((__bf16*)C)[idx] = (__bf16)acc[j][r];
        }
    }
}

// ---------------------------------------------------------------- RoPE (in-place, bf16)
__global__ void rope_kernel(__bf16* __restrict__ X, int heads, int total, float postscale) {
    int i = blockIdx.x * blockDim.x + threadIdx.x;
    if (i >= total) return;
    const int d    = i & 31;
    const int rest = i >> 5;
    const int hh   = rest % heads;
    const int tok  = rest / heads;
    const int t    = tok % T_;

    const size_t base = ((size_t)tok * heads + hh) * HD;
    float x0 = (float)X[base + d];
    float x1 = (float)X[base + d + 32];
    float inv_freq = __powf(10000.0f, -(float)d / 32.0f);
    float ang = (float)t * inv_freq;
    float s, c;
    sincosf(ang, &s, &c);            // precise: args up to ~2047
    X[base + d]      = (__bf16)((x0 * c - x1 * s) * postscale);
    X[base + d + 32] = (__bf16)((x1 * c + x0 * s) * postscale);
}

// ---------------------------------------------------------------- flash attention (GQA, causal)
// One workgroup = 4 waves = the 4 query heads of one KV group on one 16-query tile.
// K/V blocks (32 tokens) staged into LDS once via TDM, double-buffered, shared by all 4 heads.
// Q:(B,T,H,64) pre-scaled ; K:(B,T,G,64) ; VT:(B,G,HD,T) ; Out:(B,T,H*64) bf16
__global__ void attn_kernel(const __bf16* __restrict__ Q, const __bf16* __restrict__ Km,
                            const __bf16* __restrict__ VT, __bf16* __restrict__ Out) {
    __shared__ __align__(16) __bf16 sK[2][32][64];   // [buf][token][dim]   4 KiB x2
    __shared__ __align__(16) __bf16 sV[2][64][32];   // [buf][dim][token]   4 KiB x2
    __shared__ __align__(16) __bf16 sP[4][16][32];   // per-wave P staging

    const int QT = T_ / 16;
    const int qt = blockIdx.x % QT;
    const int g  = (blockIdx.x / QT) % G_;
    const int bb = blockIdx.x / (QT * G_);
    const int w    = threadIdx.x >> 5;   // wave in block = head in group
    const int lane = threadIdx.x & 31;
    const int hh   = g * 4 + w;
    const int q0   = qt * 16;
    const int m    = lane & 15, h = lane >> 4;

    // Q A-fragments (pre-scaled by 1/sqrt(HD) in RoPE)
    Frag16 aQ[2];
    const __bf16* qrow = Q + ((size_t)((bb * T_ + q0 + m) * H_ + hh)) * HD;
#pragma unroll
    for (int p = 0; p < 2; ++p) {
        aQ[p].q[0] = *(const uint4*)(qrow + p * 32 + h * 8);
        aQ[p].q[1] = *(const uint4*)(qrow + p * 32 + 16 + h * 8);
    }

    const __bf16* kbase  = Km + ((size_t)(bb * T_) * G_ + g) * HD;            // +tok*G_*HD
    const __bf16* vtbase = VT + ((size_t)bb * G_ + g) * HD * T_;              // +d*T_ + tok

    const v8f vzero = {};
    v8f oacc[4];
#pragma unroll
    for (int nt = 0; nt < 4; ++nt) oacc[nt] = vzero;
    float mrow[8], lrow[8];
#pragma unroll
    for (int r = 0; r < 8; ++r) { mrow[r] = -1e30f; lrow[r] = 0.0f; }

    const int nblk = (q0 + 16 + 31) >> 5;

    // prologue: TDM-stage block 0 into buffer 0 (wave 0 only; EXEC-independent DMA)
    if (w == 0) {
        tdm_load_2d_bf16((unsigned)(size_t)&sK[0][0][0], kbase, /*tile*/ 64, 32,
                         /*tensor*/ 64, 32, /*stride*/ (unsigned)(G_ * HD));
        tdm_load_2d_bf16((unsigned)(size_t)&sV[0][0][0], vtbase, /*tile*/ 32, 64,
                         /*tensor*/ 32, 64, /*stride*/ (unsigned)T_);
    }

    for (int blk = 0; blk < nblk; ++blk) {
        const int kv0 = blk * 32;
        const int cur = blk & 1;

        __syncthreads();   // prior reads of the other buffer are done everywhere
        if (w == 0) {
            if (blk + 1 < nblk) {   // kick next block's DMA, then wait for current (<=2 in flight)
                const int nk = kv0 + 32;
                tdm_load_2d_bf16((unsigned)(size_t)&sK[1 - cur][0][0],
                                 kbase + (size_t)nk * (G_ * HD), 64, 32, 64, 32,
                                 (unsigned)(G_ * HD));
                tdm_load_2d_bf16((unsigned)(size_t)&sV[1 - cur][0][0],
                                 vtbase + nk, 32, 64, 32, 64, (unsigned)T_);
                __builtin_amdgcn_s_wait_tensorcnt(2);
            } else {
                __builtin_amdgcn_s_wait_tensorcnt(0);
            }
        }
        __syncthreads();   // current buffer resident for all waves

        // ---- S = Q . K^T : two 16x16 tiles over the 32 staged kv tokens
        v8f s0 = vzero, s1 = vzero;
#pragma unroll
        for (int t2 = 0; t2 < 2; ++t2) {
#pragma unroll
            for (int p = 0; p < 2; ++p) {
                Frag16 bK;
                bK.q[0] = *(const uint4*)&sK[cur][t2 * 16 + m][p * 32 + h * 8];
                bK.q[1] = *(const uint4*)&sK[cur][t2 * 16 + m][p * 32 + 16 + h * 8];
                if (t2 == 0)
                    s0 = __builtin_amdgcn_wmma_f32_16x16x32_bf16(false, aQ[p].v, false, bK.v,
                                                                 (short)0, s0, false, false);
                else
                    s1 = __builtin_amdgcn_wmma_f32_16x16x32_bf16(false, aQ[p].v, false, bK.v,
                                                                 (short)0, s1, false, false);
            }
        }

        // ---- online softmax (DPP butterflies)
        wave_lds_sync();
#pragma unroll
        for (int r = 0; r < 8; ++r) {
            const int row  = q0 + r + 8 * h;
            const int col0 = kv0 + m, col1 = kv0 + 16 + m;
            float a0 = (col0 > row) ? -1e30f : s0[r];
            float a1 = (col1 > row) ? -1e30f : s1[r];

            const float tmax  = row_max16(fmaxf(a0, a1));
            const float newm  = fmaxf(mrow[r], tmax);
            const float alpha = __expf(mrow[r] - newm);
            const float p0 = __expf(a0 - newm);
            const float p1 = __expf(a1 - newm);
            const float rsum = row_sum16(p0 + p1);
            lrow[r] = lrow[r] * alpha + rsum;
            mrow[r] = newm;
#pragma unroll
            for (int nt = 0; nt < 4; ++nt) oacc[nt][r] *= alpha;

            sP[w][r + 8 * h][m]      = (__bf16)p0;
            sP[w][r + 8 * h][16 + m] = (__bf16)p1;
        }
        wave_lds_sync();

        // ---- P back in A-layout
        Frag16 aP;
        aP.q[0] = *(const uint4*)&sP[w][m][h * 8];
        aP.q[1] = *(const uint4*)&sP[w][m][16 + h * 8];

        // ---- O += P . V from the staged, token-contiguous sV
#pragma unroll
        for (int nt = 0; nt < 4; ++nt) {
            Frag16 bV;
            bV.q[0] = *(const uint4*)&sV[cur][nt * 16 + m][8 * h];
            bV.q[1] = *(const uint4*)&sV[cur][nt * 16 + m][16 + 8 * h];
            oacc[nt] = __builtin_amdgcn_wmma_f32_16x16x32_bf16(false, aP.v, false, bV.v,
                                                               (short)0, oacc[nt], false, false);
        }
    }

    // ---- normalize and store (B,T,H*64) bf16
    float invl[8];
#pragma unroll
    for (int r = 0; r < 8; ++r) invl[r] = 1.0f / lrow[r];
#pragma unroll
    for (int nt = 0; nt < 4; ++nt) {
#pragma unroll
        for (int r = 0; r < 8; ++r) {
            const int row = q0 + r + 8 * h;
            Out[(size_t)(bb * T_ + row) * DOUT + hh * HD + nt * 16 + m] =
                (__bf16)(oacc[nt][r] * invl[r]);
        }
    }
}

// ---------------------------------------------------------------- launcher
extern "C" void kernel_launch(void* const* d_in, const int* in_sizes, int n_in,
                              void* d_out, int out_size, void* d_ws, size_t ws_size,
                              hipStream_t stream) {
    (void)in_sizes; (void)n_in; (void)out_size; (void)ws_size;
    const float* x  = (const float*)d_in[0];
    const float* Wq = (const float*)d_in[1];
    const float* Wk = (const float*)d_in[2];
    const float* Wv = (const float*)d_in[3];
    const float* Wo = (const float*)d_in[4];

    char* ws = (char*)d_ws;
    __bf16* xb   = (__bf16*)(ws);                              // 8 MiB
    __bf16* WqT  = (__bf16*)(ws + (8u  << 20));                // 2 MiB   (DOUT, DIN)
    __bf16* WkT  = (__bf16*)(ws + (10u << 20));                // 0.5 MiB (G*HD, DIN)
    __bf16* WvT  = (__bf16*)(ws + (10u << 20) + (512u << 10)); // 0.5 MiB
    __bf16* WoT  = (__bf16*)(ws + (11u << 20));                // 2 MiB   (DOUT, DOUT)
    __bf16* Qb   = (__bf16*)(ws + (13u << 20));                // 8 MiB
    __bf16* Kb   = (__bf16*)(ws + (21u << 20));                // 2 MiB
    __bf16* Vb   = (__bf16*)(ws + (23u << 20));                // 2 MiB
    __bf16* VTb  = (__bf16*)(ws + (25u << 20));                // 2 MiB
    __bf16* attn = xb;                                         // reuse: x dead after QKV GEMMs

    const int thr = 256;
    f2bf_kernel  <<<(NTOK * DIN) / thr, thr, 0, stream>>>(x, xb, NTOK * DIN);
    f2bf_t_kernel<<<(DIN * DOUT) / thr, thr, 0, stream>>>(Wq, WqT, DIN, DOUT);
    f2bf_t_kernel<<<(DIN * G_ * HD) / thr, thr, 0, stream>>>(Wk, WkT, DIN, G_ * HD);
    f2bf_t_kernel<<<(DIN * G_ * HD) / thr, thr, 0, stream>>>(Wv, WvT, DIN, G_ * HD);
    f2bf_t_kernel<<<(DOUT * DOUT) / thr, thr, 0, stream>>>(Wo, WoT, DOUT, DOUT);

    {
        int waves = (NTOK / 16) * (DOUT / 64);       // 4096
        gemm_wmma_nt<false><<<waves * 32 / thr, thr, 0, stream>>>(xb, WqT, Qb, NTOK, DOUT, DIN);
    }
    {
        int waves = (NTOK / 16) * ((G_ * HD) / 64);  // 1024
        gemm_wmma_nt<false><<<waves * 32 / thr, thr, 0, stream>>>(xb, WkT, Kb, NTOK, G_ * HD, DIN);
        gemm_wmma_nt<false><<<waves * 32 / thr, thr, 0, stream>>>(xb, WvT, Vb, NTOK, G_ * HD, DIN);
    }

    rope_kernel<<<(NTOK * H_ * 32) / thr, thr, 0, stream>>>(Qb, H_, NTOK * H_ * 32, 0.125f);
    rope_kernel<<<(NTOK * G_ * 32) / thr, thr, 0, stream>>>(Kb, G_, NTOK * G_ * 32, 1.0f);
    vtrans_kernel<<<(NTOK * G_ * HD) / thr, thr, 0, stream>>>(Vb, VTb);

    // flash attention: 4 waves (heads of one group) per block, TDM-staged KV
    {
        int blocks = B_ * G_ * (T_ / 16);            // 1024 blocks x 128 threads
        attn_kernel<<<blocks, 128, 0, stream>>>(Qb, Kb, VTb, attn);
    }

    {
        int waves = (NTOK / 16) * (DOUT / 64);       // 4096
        gemm_wmma_nt<true><<<waves * 32 / thr, thr, 0, stream>>>(attn, WoT, d_out, NTOK, DOUT, DOUT);
    }
}
